// GCN_29506425323976
// MI455X (gfx1250) — compile-verified, matrix operand order checked by reading, not run on hardware
//
#include <hip/hip_runtime.h>
#include <hip/hip_bf16.h>

typedef __attribute__((ext_vector_type(16))) _Float16 v16h;
typedef __attribute__((ext_vector_type(8)))  _Float16 v8h;
typedef __attribute__((ext_vector_type(8)))  float    v8f;

#define N_NODES 50000
#define N_EDGES 800000
#define IN_F    128
#define H_F     256
#define OUT_F   64

// ---------- small utility kernels ----------

__global__ void zero_f32(float* p, int n) {
    int i = blockIdx.x * blockDim.x + threadIdx.x;
    if (i < n) p[i] = 0.0f;
}

__global__ void deg_kernel(const long long* __restrict__ ei, float* __restrict__ deg) {
    int e = blockIdx.x * blockDim.x + threadIdx.x;
    if (e < N_EDGES) {
        int d = (int)ei[N_EDGES + e];     // dst
        atomicAdd(&deg[d], 1.0f);
    }
}

__global__ void norm_kernel(float* __restrict__ deg_norm) {
    int i = blockIdx.x * blockDim.x + threadIdx.x;
    if (i < N_NODES) {
        float d = deg_norm[i];
        d = d > 0.0f ? d : 1.0f;
        deg_norm[i] = rsqrtf(d);          // deg^-1/2, in place
    }
}

// WT[n][k] = (f16) W[k][n]
__global__ void transpose_w_f16(const float* __restrict__ W, _Float16* __restrict__ WT,
                                int K, int N) {
    int t = blockIdx.x * blockDim.x + threadIdx.x;
    if (t < K * N) {
        int k = t / N, n = t % N;
        WT[n * K + k] = (_Float16)W[k * N + n];
    }
}

// xn[n, :] = h[n, :] * norm[n]   (128 feats, float4 per thread)
__global__ void scale_rows128(const float* __restrict__ h,
                              const float* __restrict__ norm,
                              float* __restrict__ xn) {
    int t = blockIdx.x * blockDim.x + threadIdx.x;   // node*32 + chunk
    int node = t >> 5, c = t & 31;
    float s = norm[node];
    float4 v = ((const float4*)h)[node * 32 + c];
    float4 o = { v.x * s, v.y * s, v.z * s, v.w * s };
    ((float4*)xn)[node * 32 + c] = o;
}

// edge-parallel scatter-add: agg[dst, :] += x[src, :]; FEATS = 4 << SHIFT
template <int SHIFT>
__global__ void edge_agg(const long long* __restrict__ ei,
                         const float* __restrict__ x,
                         float* __restrict__ agg) {
    int t = blockIdx.x * blockDim.x + threadIdx.x;
    int e = t >> SHIFT;
    int c = t & ((1 << SHIFT) - 1);
    const int F = 4 << SHIFT;
    int s = (int)ei[e];
    int d = (int)ei[N_EDGES + e];
    float4 v = *(const float4*)(x + (long long)s * F + c * 4);
    float* o = agg + (long long)d * F + c * 4;
    atomicAdd(o + 0, v.x);
    atomicAdd(o + 1, v.y);
    atomicAdd(o + 2, v.z);
    atomicAdd(o + 3, v.w);
}

__device__ __forceinline__ unsigned hashu(unsigned x) {
    x ^= x >> 16; x *= 0x7feb352du;
    x ^= x >> 15; x *= 0x846ca68bu;
    x ^= x >> 16; return x;
}

// ---------- WMMA GEMM 1: h1 = f16( dropout(relu((agg1 @ W1)*norm + b1)) * norm )
// A: [M,128] f32, WT: [256][128] f16 (W1 transposed), out: [M,256] f16
__global__ void __launch_bounds__(128)
gemm1_kernel(const float* __restrict__ A,
             const _Float16* __restrict__ WT,
             const float* __restrict__ bias,
             const float* __restrict__ norm,
             _Float16* __restrict__ out) {
    __shared__ float As[16][IN_F + 4];        // padded: conflict-free fragment reads

    const int tid   = threadIdx.x;
    const int lane  = tid & 31;
    const int wave  = tid >> 5;
    const int mbase = blockIdx.x * 16;
    const int nbase = (blockIdx.y * 4 + wave) * 16;
    const int coln  = nbase + (lane & 15);
    const int arow  = lane & 15;
    const int kl    = (lane & 16) >> 1;       // +8 K offset for lanes 16..31

    __builtin_prefetch(WT + (long long)coln * IN_F, 0, 1);

    // cooperative stage of the 16x128 f32 A tile (shared by all 4 waves)
    {
        const float4* src = (const float4*)(A + (long long)mbase * IN_F);
#pragma unroll
        for (int i = tid; i < 16 * 32; i += 128) {
            int r = i >> 5, cc = i & 31;
            *(float4*)(&As[r][cc * 4]) = src[r * 32 + cc];
        }
    }
    __syncthreads();

    v8f c = {};
#pragma unroll
    for (int kk = 0; kk < IN_F; kk += 32) {
        const int kb = kk + kl;
        float4 a0 = *(const float4*)(&As[arow][kb]);
        float4 a1 = *(const float4*)(&As[arow][kb + 4]);
        float4 a2 = *(const float4*)(&As[arow][kb + 16]);
        float4 a3 = *(const float4*)(&As[arow][kb + 20]);
        v8h  b0 = *(const v8h*)(WT + (long long)coln * IN_F + kb);
        v8h  b1 = *(const v8h*)(WT + (long long)coln * IN_F + kb + 16);
        v16h a, b;
        a[0] = (_Float16)a0.x; a[1] = (_Float16)a0.y; a[2] = (_Float16)a0.z; a[3] = (_Float16)a0.w;
        a[4] = (_Float16)a1.x; a[5] = (_Float16)a1.y; a[6] = (_Float16)a1.z; a[7] = (_Float16)a1.w;
        a[8] = (_Float16)a2.x; a[9] = (_Float16)a2.y; a[10] = (_Float16)a2.z; a[11] = (_Float16)a2.w;
        a[12] = (_Float16)a3.x; a[13] = (_Float16)a3.y; a[14] = (_Float16)a3.z; a[15] = (_Float16)a3.w;
#pragma unroll
        for (int j = 0; j < 8; ++j) { b[j] = b0[j]; b[8 + j] = b1[j]; }
        c = __builtin_amdgcn_wmma_f32_16x16x32_f16(false, a, false, b,
                                                   (short)0, c, false, false);
    }

    const int mo = (lane & 16) >> 1;          // +8 row offset for lanes 16..31
#pragma unroll
    for (int r = 0; r < 8; ++r) {
        int m = mbase + r + mo;
        float nm = norm[m];
        float v = c[r] * nm + bias[coln];
        v = v > 0.0f ? v : 0.0f;                               // relu
        unsigned idx = (unsigned)(m * H_F + coln);
        v = (hashu(idx ^ 0x2545F491u) & 1u) ? v * 2.0f : 0.0f; // dropout p=0.5
        out[m * H_F + coln] = (_Float16)(v * nm);              // fuse layer-2 pre-norm
    }
}

// ---------- WMMA GEMM 2: xw = h1f16 @ W2
// A: [M,256] f16 (already norm-scaled), WT: [64][256] f16 (W2 transposed), out: [M,64] f32
__global__ void __launch_bounds__(128)
gemm2_kernel(const _Float16* __restrict__ A,
             const _Float16* __restrict__ WT,
             float* __restrict__ out) {
    __shared__ _Float16 As[16][H_F + 8];      // padded: conflict-free fragment reads

    const int tid   = threadIdx.x;
    const int lane  = tid & 31;
    const int wave  = tid >> 5;
    const int mbase = blockIdx.x * 16;
    const int nbase = wave * 16;              // 4 waves cover N=64
    const int coln  = nbase + (lane & 15);
    const int arow  = lane & 15;
    const int kl    = (lane & 16) >> 1;

    __builtin_prefetch(WT + (long long)coln * H_F, 0, 1);

    // cooperative stage of the 16x256 f16 A tile (8 KB)
    {
        const float4* src = (const float4*)(A + (long long)mbase * H_F);
#pragma unroll
        for (int i = tid; i < 16 * 32; i += 128) {
            int r = i >> 5, cc = i & 31;
            *(float4*)(&As[r][cc * 8]) = src[r * 32 + cc];
        }
    }
    __syncthreads();

    v8f c = {};
#pragma unroll
    for (int kk = 0; kk < H_F; kk += 32) {
        const int kb = kk + kl;
        v8h a0 = *(const v8h*)(&As[arow][kb]);
        v8h a1 = *(const v8h*)(&As[arow][kb + 16]);
        v8h b0 = *(const v8h*)(WT + (long long)coln * H_F + kb);
        v8h b1 = *(const v8h*)(WT + (long long)coln * H_F + kb + 16);
        v16h a, b;
#pragma unroll
        for (int j = 0; j < 8; ++j) {
            a[j] = a0[j]; a[8 + j] = a1[j];
            b[j] = b0[j]; b[8 + j] = b1[j];
        }
        c = __builtin_amdgcn_wmma_f32_16x16x32_f16(false, a, false, b,
                                                   (short)0, c, false, false);
    }

    const int mo = (lane & 16) >> 1;
#pragma unroll
    for (int r = 0; r < 8; ++r) {
        int m = mbase + r + mo;
        out[m * OUT_F + coln] = c[r];
    }
}

// d_out[n,f] = out_agg[n,f] * norm[n] + b2[f]
__global__ void finalize_kernel(const float* __restrict__ agg,
                                const float* __restrict__ norm,
                                const float* __restrict__ b2,
                                float* __restrict__ out) {
    int t = blockIdx.x * blockDim.x + threadIdx.x;  // node*64 + f
    int n = t >> 6, f = t & 63;
    out[t] = agg[t] * norm[n] + b2[f];
}

// ---------- launch ----------

extern "C" void kernel_launch(void* const* d_in, const int* in_sizes, int n_in,
                              void* d_out, int out_size, void* d_ws, size_t ws_size,
                              hipStream_t stream) {
    const float*     h  = (const float*)d_in[0];
    const float*     W1 = (const float*)d_in[1];
    const float*     b1 = (const float*)d_in[2];
    const float*     W2 = (const float*)d_in[3];
    const float*     b2 = (const float*)d_in[4];
    const long long* ei = (const long long*)d_in[5];

    char* ws = (char*)d_ws;
    float*    norm   = (float*)(ws + 0);            // 200 KB (deg, then norm in place)
    float*    xn     = (float*)(ws + (1 << 18));    // 25.6 MB  (reused later as h1 f16)
    float*    agg1   = (float*)(ws + 26214400);     // 25.6 MB  (reused later as xw)
    float*    outagg = (float*)(ws + 52428800);     // 12.8 MB
    _Float16* w1t    = (_Float16*)(ws + 65273856);  // 64 KB   [256][128]
    _Float16* w2t    = (_Float16*)(ws + 65339392);  // 32 KB   [64][256]
    _Float16* h1     = (_Float16*)xn;               // alias: xn dead after edge_agg<5>
    float*    xw     = agg1;                        // alias: agg1 dead after gemm1

    // weight prep (f16, transposed for vectorized WMMA fragment loads)
    transpose_w_f16<<<(IN_F * H_F + 255) / 256, 256, 0, stream>>>(W1, w1t, IN_F, H_F);
    transpose_w_f16<<<(H_F * OUT_F + 255) / 256, 256, 0, stream>>>(W2, w2t, H_F, OUT_F);

    // degree -> norm
    zero_f32<<<(N_NODES + 255) / 256, 256, 0, stream>>>(norm, N_NODES);
    deg_kernel<<<(N_EDGES + 255) / 256, 256, 0, stream>>>(ei, norm);
    norm_kernel<<<(N_NODES + 255) / 256, 256, 0, stream>>>(norm);

    // layer 1: pre-scale, aggregate (atomics), GEMM + fused epilogue
    scale_rows128<<<(N_NODES * 32) / 256, 256, 0, stream>>>(h, norm, xn);
    zero_f32<<<(N_NODES * IN_F + 255) / 256, 256, 0, stream>>>(agg1, N_NODES * IN_F);
    edge_agg<5><<<(N_EDGES * 32) / 256, 256, 0, stream>>>(ei, xn, agg1);
    gemm1_kernel<<<dim3(N_NODES / 16, 4), 128, 0, stream>>>(agg1, w1t, b1, norm, h1);

    // layer 2: GEMM, aggregate (atomics), finalize
    gemm2_kernel<<<dim3(N_NODES / 16, 1), 128, 0, stream>>>(h1, w2t, xw);
    zero_f32<<<(N_NODES * OUT_F + 255) / 256, 256, 0, stream>>>(outagg, N_NODES * OUT_F);
    edge_agg<4><<<(N_EDGES * 16) / 256, 256, 0, stream>>>(ei, xw, outagg);
    finalize_kernel<<<(N_NODES * OUT_F) / 256, 256, 0, stream>>>(outagg, norm, b2, (float*)d_out);
}
